// GeometryGNN_19447611916796
// MI455X (gfx1250) — compile-verified, matrix operand order checked by reading, not run on hardware
//
#include <hip/hip_runtime.h>
#include <hip/hip_bf16.h>

// ---------------------------------------------------------------------------
// GeometryGNN forward on gfx1250 (MI455X).
// B=32, N=8192, H=128, L=3.  Rows M = B*N = 262144.
// GEMMs done with v_wmma_f32_16x16x32_f16 (f16 inputs, f32 accumulate).
// ---------------------------------------------------------------------------

typedef __attribute__((ext_vector_type(16))) _Float16 v16h;
typedef __attribute__((ext_vector_type(8)))  _Float16 v8h;
typedef __attribute__((ext_vector_type(8)))  float    v8f;
typedef __attribute__((ext_vector_type(4)))  float    v4f;

#define H128   128
#define NNODE  8192
#define MROWS  262144          // B*N
#define TM     128             // rows per block tile
#define SA     136             // LDS half-stride for A tiles (pad to dodge conflicts)
#define SW     136             // LDS half-stride for transposed weights
#define SO     136             // LDS half-stride in output kernel
#define STF    68              // float stride of t tile in output kernel
#define BN_CNT 262144.0f
#define EPSF   1e-5f

// Load a 16x32 f16 WMMA fragment (A-layout; B uses same layout from a
// transposed [n][k] tile).  Per ISA 7.12.2: lane L holds row M = L%16,
// halves[0..7]  = K = kb + (L/16)*8 + 0..7,
// halves[8..15] = K = kb + 16 + (L/16)*8 + 0..7  -> two 16B contiguous loads.
__device__ __forceinline__ v16h load_frag16(const _Float16* tile, int rowOff,
                                            int kb, int stride) {
    const int lane = threadIdx.x & 31;
    const int m    = lane & 15;
    const int ks   = (lane >> 4) << 3;
    const _Float16* p = tile + (rowOff + m) * stride + kb + ks;
    v8h lo = *(const v8h*)p;
    v8h hi = *(const v8h*)(p + 16);
    return __builtin_shufflevector(lo, hi, 0,1,2,3,4,5,6,7,
                                           8,9,10,11,12,13,14,15);
}

// ---------------------------------------------------------------------------
// h = x @ emb_w + emb_b     (x: [M,2], emb_w: [2,H], h: [M,H])
// one thread per float4 of h; 8388608 float4 total.
// ---------------------------------------------------------------------------
__global__ __launch_bounds__(256) void k_embed(const float* __restrict__ x,
                                               const float* __restrict__ ew,
                                               const float* __restrict__ eb,
                                               float* __restrict__ h) {
    const int i4 = blockIdx.x * 256 + threadIdx.x;     // < 8388608
    const int row = i4 >> 5;
    const int c4  = (i4 & 31) << 2;
    const float x0 = x[(size_t)row * 2 + 0];
    const float x1 = x[(size_t)row * 2 + 1];
    v4f w0 = *(const v4f*)(ew + c4);
    v4f w1 = *(const v4f*)(ew + 128 + c4);
    v4f b  = *(const v4f*)(eb + c4);
    v4f o;
#pragma unroll
    for (int j = 0; j < 4; ++j) o[j] = x0 * w0[j] + x1 * w1[j] + b[j];
    *(v4f*)(h + (size_t)row * H128 + c4) = o;
}

// ---------------------------------------------------------------------------
__global__ __launch_bounds__(256) void k_zero_stats(float* __restrict__ s) {
    s[threadIdx.x] = 0.0f;     // zeros gsum[128] ++ gsum2[128]
}

// ---------------------------------------------------------------------------
// Fused layer GEMM:  hnew = (h@lin_w + lin_b + (h[n-1]+h[n+1])@nb_w + nb_b)*m
// also accumulates per-channel sum / sum-of-squares for BatchNorm.
// One block = 128 rows x 128 cols.  8 waves; wave w owns rows 16w..16w+15.
// ---------------------------------------------------------------------------
__global__ __launch_bounds__(256)
void k_layer_gemm(const float* __restrict__ h, const float* __restrict__ mask,
                  const float* __restrict__ lw, const float* __restrict__ lb,
                  const float* __restrict__ nw, const float* __restrict__ nb,
                  float* __restrict__ hnew,
                  float* __restrict__ gsum, float* __restrict__ gsum2) {
    extern __shared__ char smem[];
    _Float16* aT  = (_Float16*)smem;            // [128][SA]  h tile (f16)
    _Float16* anT = aT  + TM * SA;              // [128][SA]  neighbor-sum tile
    _Float16* wlT = anT + TM * SA;              // [n][k] lin_w^T
    _Float16* wnT = wlT + H128 * SW;            // [n][k] nb_w^T
    float*    lsum  = (float*)(wnT + H128 * SW);
    float*    lsum2 = lsum + H128;

    const int t       = threadIdx.x;
    const int rowBase = blockIdx.x * TM;        // tile never straddles a batch
    const int nbase   = rowBase & (NNODE - 1);
    const int bbase   = rowBase - nbase;

    if (t < 128) lsum[t] = 0.0f; else lsum2[t - 128] = 0.0f;

    // ---- stage A tiles (f32 -> f16), neighbor sum with ring wrap ----
#pragma unroll
    for (int it = 0; it < 16; ++it) {
        int f4 = t + it * 256;                  // 0..4095 float4s
        int r  = f4 >> 5;
        int c4 = (f4 & 31) << 2;
        int n  = nbase + r;
        v4f a  = *(const v4f*)(h + (size_t)(rowBase + r) * H128 + c4);
        v4f am = *(const v4f*)(h + (size_t)(bbase + ((n + NNODE - 1) & (NNODE - 1))) * H128 + c4);
        v4f ap = *(const v4f*)(h + (size_t)(bbase + ((n + 1) & (NNODE - 1))) * H128 + c4);
        _Float16* pa  = aT  + r * SA + c4;
        _Float16* pan = anT + r * SA + c4;
#pragma unroll
        for (int j = 0; j < 4; ++j) {
            pa[j]  = (_Float16)a[j];
            pan[j] = (_Float16)(am[j] + ap[j]);
        }
    }
    // ---- stage both weight matrices transposed: wT[n][k] ----
#pragma unroll
    for (int it = 0; it < 64; ++it) {
        int idx = t + it * 256;                 // 0..16383
        int k = idx >> 7, n = idx & 127;
        wlT[n * SW + k] = (_Float16)lw[idx];
        wnT[n * SW + k] = (_Float16)nw[idx];
    }
    __syncthreads();

    const int wv     = t >> 5;
    const int lane   = t & 31;
    const int rowOff = wv << 4;                 // 16 rows per wave
    v8f acc[8];
#pragma unroll
    for (int nt = 0; nt < 8; ++nt) acc[nt] = (v8f){0,0,0,0,0,0,0,0};

#pragma unroll
    for (int kb = 0; kb < H128; kb += 32) {
        v16h a  = load_frag16(aT,  rowOff, kb, SA);
        v16h an = load_frag16(anT, rowOff, kb, SA);
#pragma unroll
        for (int nt = 0; nt < 8; ++nt) {
            v16h bl = load_frag16(wlT, nt * 16, kb, SW);
            acc[nt] = __builtin_amdgcn_wmma_f32_16x16x32_f16(
                false, a,  false, bl, (short)0, acc[nt], false, false);
            v16h bn2 = load_frag16(wnT, nt * 16, kb, SW);
            acc[nt] = __builtin_amdgcn_wmma_f32_16x16x32_f16(
                false, an, false, bn2, (short)0, acc[nt], false, false);
        }
    }

    // ---- epilogue: biases, mask, store hnew, BN partial stats ----
    const int cL   = lane & 15;
    const int mOff = (lane >> 4) << 3;          // C layout: M = r + (lane/16)*8
#pragma unroll
    for (int nt = 0; nt < 8; ++nt) {
        int   c    = nt * 16 + cL;
        float bias = lb[c] + nb[c];
        float s = 0.0f, s2 = 0.0f;
#pragma unroll
        for (int r = 0; r < 8; ++r) {
            int   grow = rowBase + rowOff + mOff + r;
            float v = (acc[nt][r] + bias) * mask[grow];
            hnew[(size_t)grow * H128 + c] = v;
            s += v; s2 += v * v;
        }
        atomicAdd(&lsum[c],  s);
        atomicAdd(&lsum2[c], s2);
    }
    __syncthreads();
    if (t < 128) {
        atomicAdd(&gsum[t],  lsum[t]);
        atomicAdd(&gsum2[t], lsum2[t]);
    }
}

// ---------------------------------------------------------------------------
__global__ __launch_bounds__(128)
void k_bn_reduce(const float* __restrict__ gsum, const float* __restrict__ gsum2,
                 const float* __restrict__ g, const float* __restrict__ b,
                 float* __restrict__ scale, float* __restrict__ shift) {
    const int c = threadIdx.x;
    float mean = gsum[c]  * (1.0f / BN_CNT);
    float var  = gsum2[c] * (1.0f / BN_CNT) - mean * mean;
    float a    = g[c] * rsqrtf(var + EPSF);
    scale[c] = a;
    shift[c] = b[c] - mean * a;
}

// h += relu(hnew*scale + shift)
__global__ __launch_bounds__(256)
void k_bn_apply(float* __restrict__ h, const float* __restrict__ hnew,
                const float* __restrict__ scale, const float* __restrict__ shift) {
    const int i4 = blockIdx.x * 256 + threadIdx.x;     // < 8388608
    const int c4 = (i4 & 31) << 2;
    v4f sc = *(const v4f*)(scale + c4);
    v4f sh = *(const v4f*)(shift + c4);
    v4f hv = *(const v4f*)(h    + (size_t)i4 * 4);
    v4f nv = *(const v4f*)(hnew + (size_t)i4 * 4);
#pragma unroll
    for (int j = 0; j < 4; ++j) {
        float v = nv[j] * sc[j] + sh[j];
        hv[j] += (v > 0.0f) ? v : 0.0f;
    }
    *(v4f*)(h + (size_t)i4 * 4) = hv;
}

// ---------------------------------------------------------------------------
// out = (relu(h@out1_w + out1_b) @ out2_w + out2_b) * mask
// block = 128 rows; stage1 via WMMA into LDS, stage2 64->2 per-lane dot.
// ---------------------------------------------------------------------------
__global__ __launch_bounds__(256)
void k_out(const float* __restrict__ h, const float* __restrict__ mask,
           const float* __restrict__ w1, const float* __restrict__ b1,
           const float* __restrict__ w2, const float* __restrict__ b2,
           float* __restrict__ out) {
    extern __shared__ char smem[];
    _Float16* hT  = (_Float16*)smem;            // [128][SO]
    _Float16* w1T = hT + TM * SO;               // [64][SO]   out1_w^T [n][k]
    float*    tT  = (float*)(w1T + 64 * SO);    // [128][STF] relu stage-1

    const int t = threadIdx.x;
    const int rowBase = blockIdx.x * TM;

#pragma unroll
    for (int it = 0; it < 16; ++it) {
        int f4 = t + it * 256;
        int r  = f4 >> 5;
        int c4 = (f4 & 31) << 2;
        v4f a = *(const v4f*)(h + (size_t)(rowBase + r) * H128 + c4);
        _Float16* p = hT + r * SO + c4;
#pragma unroll
        for (int j = 0; j < 4; ++j) p[j] = (_Float16)a[j];
    }
#pragma unroll
    for (int it = 0; it < 32; ++it) {
        int idx = t + it * 256;                 // 0..8191 ; w1: [128][64]
        int k = idx >> 6, n = idx & 63;
        w1T[n * SO + k] = (_Float16)w1[idx];
    }
    __syncthreads();

    const int wv = t >> 5, lane = t & 31, rowOff = wv << 4;
    v8f acc[4];
#pragma unroll
    for (int nt = 0; nt < 4; ++nt) acc[nt] = (v8f){0,0,0,0,0,0,0,0};
#pragma unroll
    for (int kb = 0; kb < H128; kb += 32) {
        v16h a = load_frag16(hT, rowOff, kb, SO);
#pragma unroll
        for (int nt = 0; nt < 4; ++nt) {
            v16h bfr = load_frag16(w1T, nt * 16, kb, SO);
            acc[nt] = __builtin_amdgcn_wmma_f32_16x16x32_f16(
                false, a, false, bfr, (short)0, acc[nt], false, false);
        }
    }
    const int cL = lane & 15, mOff = (lane >> 4) << 3;
#pragma unroll
    for (int nt = 0; nt < 4; ++nt) {
        int   c  = nt * 16 + cL;
        float bb = b1[c];
#pragma unroll
        for (int r = 0; r < 8; ++r) {
            float v = acc[nt][r] + bb;
            tT[(rowOff + mOff + r) * STF + c] = (v > 0.0f) ? v : 0.0f;
        }
    }
    __syncthreads();

    // stage 2: lane -> (row = rowOff + lane/2, j = lane&1)
    const int row = rowOff + (lane >> 1);
    const int j   = lane & 1;
    float o = b2[j];
#pragma unroll
    for (int c = 0; c < 64; ++c) o += tT[row * STF + c] * w2[c * 2 + j];
    const int grow = rowBase + row;
    out[(size_t)grow * 2 + j] = o * mask[grow];
}

// ---------------------------------------------------------------------------
extern "C" void kernel_launch(void* const* d_in, const int* in_sizes, int n_in,
                              void* d_out, int out_size, void* d_ws, size_t ws_size,
                              hipStream_t stream) {
    const float* x     = (const float*)d_in[0];
    const float* mask  = (const float*)d_in[1];
    const float* emb_w = (const float*)d_in[2];
    const float* emb_b = (const float*)d_in[3];
    const float* lin_w = (const float*)d_in[4];
    const float* lin_b = (const float*)d_in[5];
    const float* nb_w  = (const float*)d_in[6];
    const float* nb_b  = (const float*)d_in[7];
    const float* bn_g  = (const float*)d_in[8];
    const float* bn_b  = (const float*)d_in[9];
    const float* o1w   = (const float*)d_in[10];
    const float* o1b   = (const float*)d_in[11];
    const float* o2w   = (const float*)d_in[12];
    const float* o2b   = (const float*)d_in[13];
    float* out = (float*)d_out;

    float* h     = (float*)d_ws;                 // [262144][128] f32
    float* hnew  = h + (size_t)MROWS * H128;     // [262144][128] f32
    float* gsum  = hnew + (size_t)MROWS * H128;  // [128]
    float* gsum2 = gsum + 128;
    float* scale = gsum2 + 128;
    float* shift = scale + 128;

    const size_t smemGemm = (size_t)(2 * TM * SA + 2 * H128 * SW) * sizeof(_Float16)
                          + 256 * sizeof(float);                 // ~140 KB
    const size_t smemOut  = (size_t)(TM * SO + 64 * SO) * sizeof(_Float16)
                          + (size_t)TM * STF * sizeof(float);    // ~87 KB

    (void)hipFuncSetAttribute((const void*)k_layer_gemm,
        hipFuncAttributeMaxDynamicSharedMemorySize, (int)smemGemm);
    (void)hipFuncSetAttribute((const void*)k_out,
        hipFuncAttributeMaxDynamicSharedMemorySize, (int)smemOut);

    const int n4blocks = (MROWS * H128 / 4) / 256;   // 32768

    k_embed<<<n4blocks, 256, 0, stream>>>(x, emb_w, emb_b, h);

    for (int i = 0; i < 3; ++i) {
        k_zero_stats<<<1, 256, 0, stream>>>(gsum);   // zeros gsum+gsum2
        k_layer_gemm<<<MROWS / TM, 256, smemGemm, stream>>>(
            h, mask,
            lin_w + (size_t)i * H128 * H128, lin_b + (size_t)i * H128,
            nb_w  + (size_t)i * H128 * H128, nb_b  + (size_t)i * H128,
            hnew, gsum, gsum2);
        k_bn_reduce<<<1, 128, 0, stream>>>(gsum, gsum2,
            bn_g + (size_t)i * H128, bn_b + (size_t)i * H128, scale, shift);
        k_bn_apply<<<n4blocks, 256, 0, stream>>>(h, hnew, scale, shift);
    }

    k_out<<<MROWS / TM, 256, smemOut, stream>>>(h, mask, o1w, o1b, o2w, o2b, out);
}
